// TPUGroupedGEMMLinear_76802605187558
// MI455X (gfx1250) — matrix operand order repaired and checked by reading, not compiled, not run on hardware
//
#include <hip/hip_runtime.h>
#include <hip/hip_bf16.h>

// Problem constants (from reference): N=8192, D_IN=1024, D_OUT=4096, 8 groups.
#define NROWS     8192
#define DIN       1024
#define DOUT      4096
#define NGROUPS   8
#define TILE_M    32
#define MAX_TILES 264      // 8192/32 + 8 (padding per group)
#define KC        64       // K-chunk staged per LDS buffer

typedef __attribute__((ext_vector_type(8)))  float  v8f;
typedef __attribute__((ext_vector_type(16))) __bf16 v16bf;
typedef __attribute__((ext_vector_type(8)))  __bf16 v8bf;

// ---------------- bucketing kernels ----------------

__global__ void init_ws_kernel(int* counts, int* cursors, int* perm) {
    int i = blockIdx.x * 256 + threadIdx.x;
    if (i < NGROUPS) { counts[i] = 0; cursors[i] = 0; }
    if (i < MAX_TILES * TILE_M) perm[i] = -1;
}

__global__ void count_kernel(const int* __restrict__ gidx, int* counts) {
    int i = blockIdx.x * 256 + threadIdx.x;
    if (i < NROWS) atomicAdd(&counts[gidx[i]], 1);
}

__global__ void scan_tiles_kernel(const int* __restrict__ counts, int* offsets,
                                  int* tile_g, int* tile_b, int* ntiles) {
    if (threadIdx.x == 0 && blockIdx.x == 0) {
        int off = 0, t = 0;
        for (int g = 0; g < NGROUPS; ++g) {
            offsets[g] = off;
            int c  = counts[g];
            int nt = (c + TILE_M - 1) >> 5;                 // 32-row tiles (padded)
            for (int k = 0; k < nt; ++k) { tile_g[t] = g; tile_b[t] = off + k * TILE_M; ++t; }
            off += nt * TILE_M;
        }
        ntiles[0] = t;
    }
}

__global__ void scatter_kernel(const int* __restrict__ gidx,
                               const int* __restrict__ offsets,
                               int* cursors, int* perm) {
    int i = blockIdx.x * 256 + threadIdx.x;
    if (i < NROWS) {
        int g = gidx[i];
        int pos = offsets[g] + atomicAdd(&cursors[g], 1);
        perm[pos] = i;    // intra-group order is schedule-dependent; output is not
    }
}

// ---------------- helpers ----------------

// Direct packed convert: two f32 -> one dword of 2x bf16 (RNE).
// (The union/half-store form lowers to scalar cvt + shifts/moves; this doesn't.)
__device__ __forceinline__ unsigned pack_bf16(float a, float b) {
    unsigned r;
    asm("v_cvt_pk_bf16_f32 %0, %1, %2" : "=v"(r) : "v"(a), "v"(b));
    return r;
}

__device__ __forceinline__ float f4c(const float4& v, int j) {
    switch (j) { case 0: return v.x; case 1: return v.y; case 2: return v.z; default: return v.w; }
}

// ---------------- grouped-GEMM WMMA kernel ----------------
// Block: 256 threads (8 waves). Tile: 32 rows x 128 cols. Each wave: two
// 16x16 f32 accumulators (two M-subtiles, shared B fragment) via
// v_wmma_f32_16x16x32_bf16. K-loop over 1024 in 64-wide chunks,
// double-buffered LDS with register prefetch between barrier and WMMA burst.

__global__ __launch_bounds__(256)
void grouped_gemm_kernel(const float* __restrict__ x,      // [NROWS, DIN]
                         const float* __restrict__ w,      // [NGROUPS, DIN, DOUT]
                         const float* __restrict__ bias,   // [NGROUPS, DOUT]
                         float* __restrict__ out,          // [NROWS, DOUT]
                         const int* __restrict__ tile_g,
                         const int* __restrict__ tile_b,
                         const int* __restrict__ perm,
                         const int* __restrict__ ntiles) {
    if ((int)blockIdx.y >= ntiles[0]) return;   // uniform exit, before any barrier

    __shared__ __align__(16) __bf16 Al[2][TILE_M][72];   // 32 rows  x 64 K (stride 72)
    __shared__ __align__(16) __bf16 Bl[2][128][72];      // 128 cols x 64 K (transposed, stride 72)
    __shared__ int s_rows[TILE_M];

    const int tid   = threadIdx.x;
    const int g     = tile_g[blockIdx.y];
    const int base  = tile_b[blockIdx.y];
    const int nbase = blockIdx.x * 128;
    const float* wg = w + (size_t)g * DIN * DOUT;

    if (tid < TILE_M) s_rows[tid] = perm[base + tid];
    __syncthreads();

    // --- prefetch lane assignments ---
    const int am = tid >> 3;              // A: row 0..31
    const int ak = (tid & 7) << 3;        // A: k offset 0..56 (8 floats)
    const int bn = (tid & 31) << 2;       // B: col quad 0..124
    const int bk = (tid >> 5) << 3;       // B: k base 0..56 (8 consecutive k)
    const int arow = s_rows[am];
    const float* aptr = (arow >= 0) ? (x + (size_t)arow * DIN) : nullptr;

    float4 pa0{}, pa1{};
    float4 pb[8];

    // initial prefetch, k0 = 0
    if (aptr) { pa0 = *(const float4*)(aptr + ak); pa1 = *(const float4*)(aptr + ak + 4); }
    #pragma unroll
    for (int p = 0; p < 8; ++p)
        pb[p] = *(const float4*)(wg + (size_t)(bk + p) * DOUT + nbase + bn);

    const int lane = tid & 31;
    const int wv   = tid >> 5;            // wave id: 16-col subtile
    const int half = lane >> 4;
    const int lm   = lane & 15;
    const int bxor = (wv & 3) * 8;        // B k-offset swizzle for this wave's columns
    v8f acc0 = {}, acc1 = {};

    const int NKC = DIN / KC;             // 16 chunks
    #pragma unroll 2
    for (int kc = 0; kc < NKC; ++kc) {
        const int buf = kc & 1;

        // ---- commit prefetched chunk to LDS (fp32 -> bf16, packed stores) ----
        {
            uint4 ca;
            ca.x = pack_bf16(pa0.x, pa0.y); ca.y = pack_bf16(pa0.z, pa0.w);
            ca.z = pack_bf16(pa1.x, pa1.y); ca.w = pack_bf16(pa1.z, pa1.w);
            *(uint4*)&Al[buf][am][ak] = ca;                    // one b128 per thread
            #pragma unroll
            for (int j = 0; j < 4; ++j) {                      // transpose 4 cols x 8 k
                const int r  = bn + j;
                const int xk = ((r >> 4) & 3) * 8;             // bank-spread swizzle
                uint4 cb;
                cb.x = pack_bf16(f4c(pb[0], j), f4c(pb[1], j));
                cb.y = pack_bf16(f4c(pb[2], j), f4c(pb[3], j));
                cb.z = pack_bf16(f4c(pb[4], j), f4c(pb[5], j));
                cb.w = pack_bf16(f4c(pb[6], j), f4c(pb[7], j));
                *(uint4*)&Bl[buf][r][bk ^ xk] = cb;            // b128, 16B granule
            }
        }
        __syncthreads();

        // ---- issue next chunk's global loads (overlap with WMMA below) ----
        if (kc + 1 < NKC) {
            const int k0 = (kc + 1) * KC;
            if (aptr) { pa0 = *(const float4*)(aptr + k0 + ak);
                        pa1 = *(const float4*)(aptr + k0 + ak + 4); }
            #pragma unroll
            for (int p = 0; p < 8; ++p)
                pb[p] = *(const float4*)(wg + (size_t)(k0 + bk + p) * DOUT + nbase + bn);
        }

        // ---- 4 x v_wmma_f32_16x16x32_bf16 on this chunk ----
        #pragma unroll
        for (int s = 0; s < 2; ++s) {
            // B 32x16 bf16 layout: lane half selects K {0-7,16-23} vs {8-15,24-31}
            union { v16bf v; v8bf h[2]; } Bf;
            const __bf16* brow = &Bl[buf][wv * 16 + lm][0];
            const int L0 = s * 32 + half * 8;
            Bf.h[0] = *(const v8bf*)(brow + (L0 ^ bxor));
            Bf.h[1] = *(const v8bf*)(brow + ((L0 + 16) ^ bxor));
            // A 16x32 bf16 layout: lane half selects K {0-7,16-23} vs {8-15,24-31}
            union { v16bf v; v8bf h[2]; } Af0, Af1;
            const __bf16* ap0 = &Al[buf][lm][s * 32 + half * 8];
            const __bf16* ap1 = &Al[buf][16 + lm][s * 32 + half * 8];
            Af0.h[0] = *(const v8bf*)ap0;  Af0.h[1] = *(const v8bf*)(ap0 + 16);
            Af1.h[0] = *(const v8bf*)ap1;  Af1.h[1] = *(const v8bf*)(ap1 + 16);
            acc0 = __builtin_amdgcn_wmma_f32_16x16x32_bf16(
                false, Af0.v, false, Bf.v, (short)0, acc0, false, false);
            acc1 = __builtin_amdgcn_wmma_f32_16x16x32_bf16(
                false, Af1.v, false, Bf.v, (short)0, acc1, false, false);
        }
    }

    // ---- store: C/D layout -> lane holds col lm, rows half*8 + r ----
    const int col = nbase + wv * 16 + lm;
    const float bv = bias[g * DOUT + col];
    #pragma unroll
    for (int r = 0; r < 8; ++r) {
        int ml  = half * 8 + r;
        int row = s_rows[ml];
        if (row >= 0) out[(size_t)row * DOUT + col] = acc0[r] + bv;
    }
    #pragma unroll
    for (int r = 0; r < 8; ++r) {
        int ml  = 16 + half * 8 + r;
        int row = s_rows[ml];
        if (row >= 0) out[(size_t)row * DOUT + col] = acc1[r] + bv;
    }
}

// ---------------- launch ----------------

extern "C" void kernel_launch(void* const* d_in, const int* in_sizes, int n_in,
                              void* d_out, int out_size, void* d_ws, size_t ws_size,
                              hipStream_t stream) {
    const float* x    = (const float*)d_in[0];   // inputs  [8192,1024] f32
    const int*   gidx = (const int*)  d_in[1];   // group_indices [8192] i32
    const float* w    = (const float*)d_in[2];   // w [8,1024,4096] f32
    const float* bias = (const float*)d_in[3];   // b [8,4096] f32
    float* out = (float*)d_out;

    // workspace layout (ints): counts[8] cursors[8] offsets[8] ntiles[1->pad 8]
    //                          tile_g[264] tile_b[264] perm[264*32]
    int* ws      = (int*)d_ws;
    int* counts  = ws;
    int* cursors = ws + 8;
    int* offsets = ws + 16;
    int* ntiles  = ws + 24;
    int* tile_g  = ws + 32;
    int* tile_b  = ws + 32 + MAX_TILES;
    int* perm    = ws + 32 + 2 * MAX_TILES;     // ~36 KB total

    init_ws_kernel<<<(MAX_TILES * TILE_M + 255) / 256, 256, 0, stream>>>(counts, cursors, perm);
    count_kernel<<<NROWS / 256, 256, 0, stream>>>(gidx, counts);
    scan_tiles_kernel<<<1, 64, 0, stream>>>(counts, offsets, tile_g, tile_b, ntiles);
    scatter_kernel<<<NROWS / 256, 256, 0, stream>>>(gidx, offsets, cursors, perm);

    dim3 grid(DOUT / 128, MAX_TILES);
    grouped_gemm_kernel<<<grid, 256, 0, stream>>>(x, w, bias, out,
                                                  tile_g, tile_b, perm, ntiles);
}
